// SAKEInteraction_19292993094311
// MI455X (gfx1250) — matrix operand order repaired and checked by reading, not compiled, not run on hardware
//
#include <hip/hip_runtime.h>
#include <hip/hip_bf16.h>

// ---------------------------------------------------------------------------
// SAKE interaction layer for MI455X (gfx1250, wave32).
// Memory-bound (~1GB traffic, ~22 GFLOP): fp32 data movement, all dense GEMMs
// lowered to v_wmma_f32_16x16x32_bf16 (bf16 A/B, f32 accumulate).
// A-tiles are staged in LDS *already in WMMA A-fragment bf16 layout* so each
// fragment reload is one 32B ds_load per lane; fragments are hoisted across
// the N-tile sweep (loaded once per K-step, reused for all output tiles).
// ---------------------------------------------------------------------------

typedef __attribute__((ext_vector_type(16))) __bf16 v16bf;
typedef __attribute__((ext_vector_type(8)))  float  v8f;

#define DEV static __device__ __forceinline__

constexpr float F_EPS  = 1e-8f;
constexpr float MU0    = 0.60653065971263342f;   // exp(-0.5)
constexpr float MUSTEP = (1.0f - MU0) / 19.0f;   // linspace step, R=20
constexpr float BETA   = 645.9191706f;           // (0.1*(1-exp(-0.5)))^-2

DEV unsigned short f2bf(float f) {
  unsigned int u = __float_as_uint(f);
  unsigned int r = (u + 0x7FFFu + ((u >> 16) & 1u)) >> 16;   // RNE
  return (unsigned short)r;
}

DEV v8f wmma_bf16(v16bf a, v16bf b, v8f c) {
  return __builtin_amdgcn_wmma_f32_16x16x32_bf16(false, a, false, b, (short)0, c,
                                                 false, false);
}

// Store one f32 value (row m, col k) into an LDS A-fragment bf16 buffer.
// A layout (16x32 per k-tile): lane<16 holds row=lane K={0..7,16..23},
// lane>=16 K={8..15,24..31}; element i -> k = ((i&8)<<1) + 8*(lane>=16) + (i&7).
DEV void st_afrag(unsigned short* base, int m, int k, float f) {
  const int kk   = k & 31;
  const int lane = m + ((kk & 8) ? 16 : 0);
  const int i    = ((kk & 16) >> 1) + (kk & 7);
  base[((k >> 5) * 32 + lane) * 16 + i] = f2bf(f);
}

// GEMM: A fragments (KT k-tiles) loaded once from LDS, swept over NT output
// tiles against prepped B fragments; epilogue lambda gets (nt, 16x16 f32 tile).
template <int KT, int NT, typename Epi>
DEV void gemm(const unsigned short* aF, const unsigned short* wfrag, int lane,
              Epi&& epi) {
  const v16bf* ap = reinterpret_cast<const v16bf*>(aF);
  const v16bf* bp = reinterpret_cast<const v16bf*>(wfrag);
  v16bf a[KT];
#pragma unroll
  for (int kt = 0; kt < KT; ++kt) a[kt] = ap[kt * 32 + lane];
#pragma unroll
  for (int nt = 0; nt < NT; ++nt) {
    v8f acc = {0.f, 0.f, 0.f, 0.f, 0.f, 0.f, 0.f, 0.f};
#pragma unroll
    for (int kt = 0; kt < KT; ++kt)
      acc = wmma_bf16(a[kt], bp[(kt * NT + nt) * 32 + lane], acc);
    epi(nt, acc);
  }
}

DEV float silu_f(float x) { return x / (1.f + __expf(-x)); }
DEV int   f2ord(float f)  { int b = __float_as_int(f); return b >= 0 ? b : (b ^ 0x7FFFFFFF); }
DEV float ord2f(int i)    { return __int_as_float(i >= 0 ? i : (i ^ 0x7FFFFFFF)); }

// ---------------------------------------------------------------------------
// Weight fragmentizer: W[K][N] fp32 -> bf16 fragments in WMMA B layout.
// B layout (32x16, 8 VGPRs): lanes 0-15 hold K={0..7,16..23}, lanes 16-31 K={8..15,24..31}.
// ---------------------------------------------------------------------------
__global__ void k_prep(const float* __restrict__ W, unsigned short* __restrict__ dst,
                       int K, int N, int KT, int NT) {
  const int t = blockIdx.x * blockDim.x + threadIdx.x;
  const int total = KT * NT * 512;
  if (t >= total) return;
  const int i    = t & 15;
  const int lane = (t >> 4) & 31;
  const int fidx = t >> 9;
  const int nt   = fidx % NT;
  const int kt   = fidx / NT;
  const int k = kt * 32 + ((i & 8) << 1) + ((lane >= 16) ? 8 : 0) + (i & 7);
  const int n = nt * 16 + (lane & 15);
  const float v = (k < K && n < N) ? W[k * N + n] : 0.0f;
  dst[t] = f2bf(v);
}

__global__ void k_init(int* __restrict__ segmax, float* __restrict__ denom,
                       float* __restrict__ counts, float* __restrict__ hsem,
                       float* __restrict__ comb, int n) {
  const long long t = (long long)blockIdx.x * blockDim.x + threadIdx.x;
  if (t < (long long)n * 128) hsem[t] = 0.f;
  if (t < (long long)n * 96)  comb[t] = 0.f;
  if (t < (long long)n * 4)   { segmax[t] = (int)0x80000000; denom[t] = 0.f; }
  if (t < n)                  counts[t] = 0.f;
}

// ---------------------------------------------------------------------------
// Edge pass A: geometry, RBF, edge MLP (3 WMMA GEMMs), logits + segment max.
// One wave = one 16-edge tile; 2 waves/block, per-wave LDS slices.
// ---------------------------------------------------------------------------
constexpr int HA_EF   = 5 * 512;              // e_in frags, K=160
constexpr int HA_H1   = 2 * 512;              // h1 frags, K=64
constexpr int HA_SLOT = HA_EF + HA_H1;        // halves per wave
constexpr int FA_SLOT = 512 + 320 + 16;       // hE f32 + rbf f32 + iiB

__global__ void __launch_bounds__(64) k_edge_a(
    const float* __restrict__ h, const float* __restrict__ x,
    const int* __restrict__ idx_i, const int* __restrict__ idx_j,
    const float* __restrict__ b_edge_in, const float* __restrict__ b_edge_h,
    const float* __restrict__ b_edge_out,
    const float* __restrict__ W_att, const float* __restrict__ b_att,
    const unsigned short* __restrict__ f_ein, const unsigned short* __restrict__ f_eh,
    const unsigned short* __restrict__ f_eout,
    float* __restrict__ hedge_ws, float* __restrict__ logit_ws,
    int* __restrict__ segmax_ws, int Enum) {
  __shared__ __attribute__((aligned(32))) unsigned short sbf[2 * HA_SLOT];
  __shared__ float sfl[2 * FA_SLOT];
  const int wave = threadIdx.x >> 5;
  const int lane = threadIdx.x & 31;
  unsigned short* eF  = sbf + wave * HA_SLOT;  // A frags: h_i|h_j|filtered|d|pad0
  unsigned short* h1F = eF + HA_EF;            // A frags: hidden 64
  float* hE  = sfl + wave * FA_SLOT;           // [16][32] f32 (logits need f32)
  float* rb  = hE + 512;                       // [16][20]
  int*   iiB = (int*)(rb + 320);               // [16]

  const int tile = blockIdx.x * 2 + wave;
  const int e0 = tile * 16;

  if (lane < 16) {
    const int m = lane;
    const int e  = e0 + m;
    const int ec = (e < Enum) ? e : (Enum - 1);
    const int ii = idx_i[ec];
    const int jj = idx_j[ec];
    iiB[m] = (e < Enum) ? ii : -1;
    const float rx = x[jj * 3 + 0] - x[ii * 3 + 0];
    const float ry = x[jj * 3 + 1] - x[ii * 3 + 1];
    const float rz = x[jj * 3 + 2] - x[ii * 3 + 2];
    const float d = sqrtf(rx * rx + ry * ry + rz * rz + F_EPS);
    st_afrag(eF, m, 148, d);   // d / SCALE, SCALE=1
#pragma unroll
    for (int c = 149; c < 160; ++c) st_afrag(eF, m, c, 0.f);
    const float ed = __expf(-d);
#pragma unroll
    for (int k = 0; k < 20; ++k) {
      const float t = ed - (MU0 + MUSTEP * (float)k);
      rb[m * 20 + k] = __expf(-BETA * t * t);
    }
  }
  __builtin_amdgcn_wave_barrier();

  {  // gather h rows cooperatively: lane<16 -> h[idx_i] (k 0..63),
     //                              lane>=16 -> h[idx_j] (k 64..127)
    const int m  = lane & 15;
    const int e  = e0 + m;
    const int ec = (e < Enum) ? e : (Enum - 1);
    const int node = (lane < 16) ? idx_i[ec] : idx_j[ec];
    const int kb   = (lane < 16) ? 0 : 64;
    const float4* h4 = reinterpret_cast<const float4*>(h);
#pragma unroll
    for (int t = 0; t < 16; ++t) {
      const float4 vv = h4[(size_t)node * 16 + t];
      st_afrag(eF, m, kb + t * 4 + 0, vv.x);
      st_afrag(eF, m, kb + t * 4 + 1, vv.y);
      st_afrag(eF, m, kb + t * 4 + 2, vv.z);
      st_afrag(eF, m, kb + t * 4 + 3, vv.w);
    }
  }
  __builtin_amdgcn_wave_barrier();

  // filtered = rbf(d) * (h_cat @ W_edge_in + b)   [K=128 -> N=20 (pad32)]
  gemm<4, 2>(eF, f_ein, lane, [&](int nt, v8f acc) {
    const int n  = nt * 16 + (lane & 15);
    const int mh = (lane >> 4) << 3;
    if (n < 20) {
      const float bn = b_edge_in[n];
#pragma unroll
      for (int r = 0; r < 8; ++r) {
        const int m = mh + r;
        st_afrag(eF, m, 128 + n, rb[m * 20 + n] * (acc[r] + bn));
      }
    }
  });
  __builtin_amdgcn_wave_barrier();

  // h1 = silu(e_in @ W_edge_h + b)   [K=160 -> N=64]
  gemm<5, 4>(eF, f_eh, lane, [&](int nt, v8f acc) {
    const int n  = nt * 16 + (lane & 15);
    const int mh = (lane >> 4) << 3;
    const float bn = b_edge_h[n];
#pragma unroll
    for (int r = 0; r < 8; ++r) st_afrag(h1F, mh + r, n, silu_f(acc[r] + bn));
  });
  __builtin_amdgcn_wave_barrier();

  // h_ij_edge = h1 @ W_edge_out + b   [K=64 -> N=32]
  gemm<2, 2>(h1F, f_eout, lane, [&](int nt, v8f acc) {
    const int n  = nt * 16 + (lane & 15);
    const int mh = (lane >> 4) << 3;
    const float bn = b_edge_out[n];
#pragma unroll
    for (int r = 0; r < 8; ++r) {
      const int m = mh + r;
      const float vv = acc[r] + bn;
      hE[m * 32 + n] = vv;
      const int e = e0 + m;
      if (e < Enum) hedge_ws[(size_t)e * 32 + n] = vv;
    }
  });
  __builtin_amdgcn_wave_barrier();

  // logits = celu(h_edge @ W_att + b, alpha=2); atomic segment max (ordered int)
  if (lane < 16) {
    const int m = lane;
    const int e = e0 + m;
    if (e < Enum) {
      const int ii = iiB[m];
#pragma unroll
      for (int hh = 0; hh < 4; ++hh) {
        float a = b_att[hh];
        for (int c = 0; c < 32; ++c) a += hE[m * 32 + c] * W_att[c * 4 + hh];
        const float lg = (a > 0.f) ? a : 2.f * (__expf(0.5f * a) - 1.f);
        logit_ws[(size_t)e * 4 + hh] = lg;
        atomicMax(&segmax_ws[ii * 4 + hh], f2ord(lg));
      }
    }
  }
}

// ---------------------------------------------------------------------------
// Edge pass B: exp(logit - segmax), denom & degree accumulation.
// ---------------------------------------------------------------------------
__global__ void k_edge_b(const int* __restrict__ idx_i, const int* __restrict__ segmax_ws,
                         float* __restrict__ logit_ws, float* __restrict__ denom_ws,
                         float* __restrict__ counts_ws, int Enum) {
  const int e = blockIdx.x * blockDim.x + threadIdx.x;
  if (e >= Enum) return;
  const int ii = idx_i[e];
#pragma unroll
  for (int hh = 0; hh < 4; ++hh) {
    const float lg = logit_ws[(size_t)e * 4 + hh];
    const float mx = ord2f(segmax_ws[ii * 4 + hh]);
    const float ex = __expf(lg - mx);
    logit_ws[(size_t)e * 4 + hh] = ex;
    atomicAdd(&denom_ws[ii * 4 + hh], ex);
  }
  atomicAdd(&counts_ws[ii], 1.0f);
}

// ---------------------------------------------------------------------------
// Edge pass C: att, semantic outer product + scatter, mix=tanh(sem@W_x_mix),
// combinations scatter. dv is folded into node pass via w_v_mix . comb_mean.
// ---------------------------------------------------------------------------
constexpr int HC_SEM  = 4 * 512;              // sem frags, K=128
constexpr int FC_SLOT = 16 * 128 + 64 + 16;   // sem f32 + dirB + iiB

__global__ void __launch_bounds__(64) k_edge_c(
    const float* __restrict__ x, const int* __restrict__ idx_i,
    const int* __restrict__ idx_j,
    const float* __restrict__ hedge_ws, const float* __restrict__ expl_ws,
    const float* __restrict__ denom_ws, const unsigned short* __restrict__ f_xmix,
    float* __restrict__ hsem_ws, float* __restrict__ comb_ws, int Enum) {
  __shared__ __attribute__((aligned(32))) unsigned short sbf[2 * HC_SEM];
  __shared__ float sfl[2 * FC_SLOT];
  const int wave = threadIdx.x >> 5;
  const int lane = threadIdx.x & 31;
  unsigned short* semF = sbf + wave * HC_SEM;  // A frags, K=128
  float* sem  = sfl + wave * FC_SLOT;          // [16][128] f32 (atomic scatter)
  float* dirB = sem + 16 * 128;                // [16][4]
  int*   iiB  = (int*)(dirB + 64);             // [16]

  const int tile = blockIdx.x * 2 + wave;
  const int e0 = tile * 16;

  if (lane < 16) {
    const int m  = lane;
    const int e  = e0 + m;
    const int ec = (e < Enum) ? e : (Enum - 1);
    const int ii = idx_i[ec];
    const int jj = idx_j[ec];
    iiB[m] = (e < Enum) ? ii : -1;
    const float rx = x[jj * 3 + 0] - x[ii * 3 + 0];
    const float ry = x[jj * 3 + 1] - x[ii * 3 + 1];
    const float rz = x[jj * 3 + 2] - x[ii * 3 + 2];
    const float d   = sqrtf(rx * rx + ry * ry + rz * rz + F_EPS);
    const float inv = 1.f / (d + F_EPS);
    dirB[m * 4 + 0] = rx * inv; dirB[m * 4 + 1] = ry * inv; dirB[m * 4 + 2] = rz * inv;
    float att[4];
#pragma unroll
    for (int hh = 0; hh < 4; ++hh)
      att[hh] = expl_ws[(size_t)ec * 4 + hh] / (denom_ws[ii * 4 + hh] + F_EPS);
    const float4* he4 = reinterpret_cast<const float4*>(hedge_ws + (size_t)ec * 32);
#pragma unroll
    for (int q = 0; q < 8; ++q) {
      const float4 vv = he4[q];
      const int c = q * 4;
#pragma unroll
      for (int hh = 0; hh < 4; ++hh) {
        const int cb = hh * 32 + c;
        const float s0 = att[hh] * vv.x, s1 = att[hh] * vv.y;
        const float s2 = att[hh] * vv.z, s3 = att[hh] * vv.w;
        sem[m * 128 + cb + 0] = s0; st_afrag(semF, m, cb + 0, s0);
        sem[m * 128 + cb + 1] = s1; st_afrag(semF, m, cb + 1, s1);
        sem[m * 128 + cb + 2] = s2; st_afrag(semF, m, cb + 2, s2);
        sem[m * 128 + cb + 3] = s3; st_afrag(semF, m, cb + 3, s3);
      }
    }
  }
  __builtin_amdgcn_wave_barrier();

  // scatter-add h_i_semantic
  for (int m = 0; m < 16; ++m) {
    const int ii = iiB[m];
    if (ii >= 0) {
      for (int c = lane; c < 128; c += 32)
        atomicAdd(&hsem_ws[(size_t)ii * 128 + c], sem[m * 128 + c]);
    }
  }

  // mix = tanh(sem @ W_x_mix); combinations scatter   [K=128 -> N=32]
  gemm<4, 2>(semF, f_xmix, lane, [&](int nt, v8f acc) {
    const int n  = nt * 16 + (lane & 15);
    const int mh = (lane >> 4) << 3;
#pragma unroll
    for (int r = 0; r < 8; ++r) {
      const int m = mh + r;
      const int ii = iiB[m];
      if (ii >= 0) {
        const float t = tanhf(acc[r]);
        atomicAdd(&comb_ws[(size_t)ii * 96 + n * 3 + 0], t * dirB[m * 4 + 0]);
        atomicAdd(&comb_ws[(size_t)ii * 96 + n * 3 + 1], t * dirB[m * 4 + 1]);
        atomicAdd(&comb_ws[(size_t)ii * 96 + n * 3 + 2], t * dirB[m * 4 + 2]);
      }
    }
  });
}

// ---------------------------------------------------------------------------
// Node pass: spatial MLP, node MLP (224->128->64), gate MLP, outputs.
// One wave = 16 nodes; all GEMMs on bf16-fragment LDS tiles.
// ---------------------------------------------------------------------------
constexpr int HD_NIN  = 7 * 512;   // node_in frags, K=224
constexpr int HD_NORM = 1 * 512;   // comb_norm2 frags, K=32
constexpr int HD_SP1  = 2 * 512;   // spatial hidden frags, K=64
constexpr int HD_T1   = 4 * 512;   // node hidden frags, K=128
constexpr int HD_HU   = 2 * 512;   // h_upd frags, K=64
constexpr int HD_SLOT = HD_NIN + HD_NORM + HD_SP1 + HD_T1 + HD_HU;  // 8192 halves
constexpr int FD_SLOT = 16 * 64 + 16 * 32;   // hRes f32 + g1 f32

__global__ void __launch_bounds__(64) k_node(
    const float* __restrict__ h, const float* __restrict__ x,
    const float* __restrict__ v,
    const float* __restrict__ counts_ws, const float* __restrict__ hsem_ws,
    const float* __restrict__ comb_ws,
    const float* __restrict__ b_node1, const float* __restrict__ b_node2,
    const float* __restrict__ b_post1, const float* __restrict__ b_post2,
    const float* __restrict__ b_vel1, const float* __restrict__ W_vel2,
    const float* __restrict__ w_v_mix,
    const unsigned short* __restrict__ f_post1, const unsigned short* __restrict__ f_post2,
    const unsigned short* __restrict__ f_node1, const unsigned short* __restrict__ f_node2,
    const unsigned short* __restrict__ f_vel1,
    float* __restrict__ out, int Nn) {
  __shared__ __attribute__((aligned(32))) unsigned short sbf[2 * HD_SLOT];
  __shared__ float sfl[2 * FD_SLOT];
  const int wave = threadIdx.x >> 5;
  const int lane = threadIdx.x & 31;
  unsigned short* nInF  = sbf + wave * HD_SLOT;  // h | h_sem | h_spatial (K=224)
  unsigned short* normF = nInF + HD_NIN;
  unsigned short* sp1F  = normF + HD_NORM;
  unsigned short* t1F   = sp1F + HD_SP1;
  unsigned short* hUF   = t1F + HD_T1;
  float* hRes = sfl + wave * FD_SLOT;            // [16][64] f32 residual
  float* g1   = hRes + 16 * 64;                  // [16][32] f32 gate hidden

  const int tile = blockIdx.x * 2 + wave;
  const int u0 = tile * 16;

  float dvx = 0.f, dvy = 0.f, dvz = 0.f;
  if (lane < 16) {
    const int m  = lane;
    const int u  = u0 + m;
    const int uc = (u < Nn) ? u : (Nn - 1);
    const float c   = counts_ws[uc];
    const float cnt = (c < 1.f) ? 1.f : c;
    const float inv = 1.f / cnt;
    for (int n = 0; n < 32; ++n) {
      const float c0 = comb_ws[(size_t)uc * 96 + n * 3 + 0] * inv;
      const float c1 = comb_ws[(size_t)uc * 96 + n * 3 + 1] * inv;
      const float c2 = comb_ws[(size_t)uc * 96 + n * 3 + 2] * inv;
      st_afrag(normF, m, n, c0 * c0 + c1 * c1 + c2 * c2);
      const float wv = w_v_mix[n];
      dvx += wv * c0; dvy += wv * c1; dvz += wv * c2;   // dv = w . comb_mean
    }
  }
  __builtin_amdgcn_wave_barrier();

  {  // stage node_in cols 0..191 cooperatively (bf16 frag layout + f32 residual)
    const int m  = lane & 15;
    const int u  = u0 + m;
    const int uc = (u < Nn) ? u : (Nn - 1);
    if (lane < 16) {
      const float4* h4 = reinterpret_cast<const float4*>(h + (size_t)uc * 64);
#pragma unroll
      for (int t = 0; t < 16; ++t) {
        const float4 vv = h4[t];
        const int k = t * 4;
        hRes[m * 64 + k + 0] = vv.x; st_afrag(nInF, m, k + 0, vv.x);
        hRes[m * 64 + k + 1] = vv.y; st_afrag(nInF, m, k + 1, vv.y);
        hRes[m * 64 + k + 2] = vv.z; st_afrag(nInF, m, k + 2, vv.z);
        hRes[m * 64 + k + 3] = vv.w; st_afrag(nInF, m, k + 3, vv.w);
      }
    } else {
      const float4* s4 = reinterpret_cast<const float4*>(hsem_ws + (size_t)uc * 128);
#pragma unroll
      for (int t = 0; t < 32; ++t) {
        const float4 vv = s4[t];
        const int k = 64 + t * 4;
        st_afrag(nInF, m, k + 0, vv.x);
        st_afrag(nInF, m, k + 1, vv.y);
        st_afrag(nInF, m, k + 2, vv.z);
        st_afrag(nInF, m, k + 3, vv.w);
      }
    }
  }
  __builtin_amdgcn_wave_barrier();

  // spatial MLP: silu(silu(norm @ W_post1 + b) @ W_post2 + b)
  gemm<1, 4>(normF, f_post1, lane, [&](int nt, v8f acc) {
    const int n  = nt * 16 + (lane & 15);
    const int mh = (lane >> 4) << 3;
    const float bn = b_post1[n];
#pragma unroll
    for (int r = 0; r < 8; ++r) st_afrag(sp1F, mh + r, n, silu_f(acc[r] + bn));
  });
  __builtin_amdgcn_wave_barrier();
  gemm<2, 2>(sp1F, f_post2, lane, [&](int nt, v8f acc) {
    const int n  = nt * 16 + (lane & 15);
    const int mh = (lane >> 4) << 3;
    const float bn = b_post2[n];
#pragma unroll
    for (int r = 0; r < 8; ++r)
      st_afrag(nInF, mh + r, 192 + n, silu_f(acc[r] + bn));
  });
  __builtin_amdgcn_wave_barrier();

  // node MLP: t1 = silu(node_in @ W_node1 + b)   [K=224 -> N=128]
  gemm<7, 8>(nInF, f_node1, lane, [&](int nt, v8f acc) {
    const int n  = nt * 16 + (lane & 15);
    const int mh = (lane >> 4) << 3;
    const float bn = b_node1[n];
#pragma unroll
    for (int r = 0; r < 8; ++r) st_afrag(t1F, mh + r, n, silu_f(acc[r] + bn));
  });
  __builtin_amdgcn_wave_barrier();

  // h_upd = h + silu(t1 @ W_node2 + b)   [K=128 -> N=64]
  gemm<4, 4>(t1F, f_node2, lane, [&](int nt, v8f acc) {
    const int n  = nt * 16 + (lane & 15);
    const int mh = (lane >> 4) << 3;
    const float bn = b_node2[n];
#pragma unroll
    for (int r = 0; r < 8; ++r) {
      const int m = mh + r;
      const float hu = hRes[m * 64 + n] + silu_f(acc[r] + bn);
      st_afrag(hUF, m, n, hu);
      const int u = u0 + m;
      if (u < Nn) out[(size_t)u * 64 + n] = hu;
    }
  });
  __builtin_amdgcn_wave_barrier();

  // gate MLP: silu(h_upd @ W_vel1 + b)   [K=64 -> N=32]
  gemm<2, 2>(hUF, f_vel1, lane, [&](int nt, v8f acc) {
    const int n  = nt * 16 + (lane & 15);
    const int mh = (lane >> 4) << 3;
    const float bn = b_vel1[n];
#pragma unroll
    for (int r = 0; r < 8; ++r) g1[(mh + r) * 32 + n] = silu_f(acc[r] + bn);
  });
  __builtin_amdgcn_wave_barrier();

  if (lane < 16) {
    const int m = lane;
    const int u = u0 + m;
    if (u < Nn) {
      float g = 0.f;
      for (int n = 0; n < 32; ++n) g += g1[m * 32 + n] * W_vel2[n];
      const float gate = 2.f / (1.f + __expf(-g));
      const float vux = gate * v[u * 3 + 0] + dvx;
      const float vuy = gate * v[u * 3 + 1] + dvy;
      const float vuz = gate * v[u * 3 + 2] + dvz;
      float* outx = out + (size_t)Nn * 64;
      float* outv = outx + (size_t)Nn * 3;
      outx[u * 3 + 0] = x[u * 3 + 0] + vux;
      outx[u * 3 + 1] = x[u * 3 + 1] + vuy;
      outx[u * 3 + 2] = x[u * 3 + 2] + vuz;
      outv[u * 3 + 0] = vux; outv[u * 3 + 1] = vuy; outv[u * 3 + 2] = vuz;
    }
  }
}

// ---------------------------------------------------------------------------
extern "C" void kernel_launch(void* const* d_in, const int* in_sizes, int n_in,
                              void* d_out, int out_size, void* d_ws, size_t ws_size,
                              hipStream_t stream) {
  const float* h         = (const float*)d_in[0];
  const float* x         = (const float*)d_in[1];
  const float* v         = (const float*)d_in[2];
  const int*   idx_i     = (const int*)d_in[3];
  const int*   idx_j     = (const int*)d_in[4];
  const float* W_edge_in = (const float*)d_in[5];
  const float* b_edge_in = (const float*)d_in[6];
  const float* W_edge_h  = (const float*)d_in[7];
  const float* b_edge_h  = (const float*)d_in[8];
  const float* W_edge_out= (const float*)d_in[9];
  const float* b_edge_out= (const float*)d_in[10];
  const float* W_att     = (const float*)d_in[11];
  const float* b_att     = (const float*)d_in[12];
  const float* W_x_mix   = (const float*)d_in[13];
  const float* W_node1   = (const float*)d_in[14];
  const float* b_node1   = (const float*)d_in[15];
  const float* W_node2   = (const float*)d_in[16];
  const float* b_node2   = (const float*)d_in[17];
  const float* W_post1   = (const float*)d_in[18];
  const float* b_post1   = (const float*)d_in[19];
  const float* W_post2   = (const float*)d_in[20];
  const float* b_post2   = (const float*)d_in[21];
  const float* W_vel1    = (const float*)d_in[22];
  const float* b_vel1    = (const float*)d_in[23];
  const float* W_vel2    = (const float*)d_in[24];
  const float* w_v_mix   = (const float*)d_in[25];

  const int Nn   = in_sizes[0] / 64;
  const int Enum = in_sizes[3];

  char* ws = (char*)d_ws;
  size_t off = 0;
  auto carve = [&](size_t bytes) -> char* {
    off = (off + 255) & ~(size_t)255;
    char* p = ws + off;
    off += bytes;
    return p;
  };
  auto fragBytes = [](int K, int N) -> size_t {
    return (size_t)((K + 31) / 32) * ((N + 15) / 16) * 512 * 2;
  };

  unsigned short* f_ein  = (unsigned short*)carve(fragBytes(128, 32));
  unsigned short* f_eh   = (unsigned short*)carve(fragBytes(160, 64));
  unsigned short* f_eout = (unsigned short*)carve(fragBytes(64, 32));
  unsigned short* f_xmix = (unsigned short*)carve(fragBytes(128, 32));
  unsigned short* f_n1   = (unsigned short*)carve(fragBytes(224, 128));
  unsigned short* f_n2   = (unsigned short*)carve(fragBytes(128, 64));
  unsigned short* f_p1   = (unsigned short*)carve(fragBytes(32, 64));
  unsigned short* f_p2   = (unsigned short*)carve(fragBytes(64, 32));
  unsigned short* f_v1   = (unsigned short*)carve(fragBytes(64, 32));

  float* hedge_ws  = (float*)carve((size_t)Enum * 32 * 4);
  float* logit_ws  = (float*)carve((size_t)Enum * 4 * 4);
  int*   segmax_ws = (int*)carve((size_t)Nn * 4 * 4);
  float* denom_ws  = (float*)carve((size_t)Nn * 4 * 4);
  float* counts_ws = (float*)carve((size_t)Nn * 4);
  float* hsem_ws   = (float*)carve((size_t)Nn * 128 * 4);
  float* comb_ws   = (float*)carve((size_t)Nn * 96 * 4);
  (void)ws_size; (void)n_in; (void)out_size;

  // weight fragmentization (deterministic, re-run every call)
  struct PF { const float* W; unsigned short* dst; int K, N; };
  const PF pf[9] = {
      {W_edge_in,  f_ein, 128, 20},
      {W_edge_h,   f_eh,  149, 64},
      {W_edge_out, f_eout, 64, 32},
      {W_x_mix,    f_xmix,128, 32},
      {W_node1,    f_n1,  224,128},
      {W_node2,    f_n2,  128, 64},
      {W_post1,    f_p1,   32, 64},
      {W_post2,    f_p2,   64, 32},
      {W_vel1,     f_v1,   64, 32},
  };
  for (int i = 0; i < 9; ++i) {
    const int KT = (pf[i].K + 31) / 32, NT = (pf[i].N + 15) / 16;
    const int total = KT * NT * 512;
    k_prep<<<(total + 255) / 256, 256, 0, stream>>>(pf[i].W, pf[i].dst,
                                                    pf[i].K, pf[i].N, KT, NT);
  }

  {  // zero/seed accumulators
    const long long total = (long long)Nn * 128;
    k_init<<<(int)((total + 255) / 256), 256, 0, stream>>>(
        segmax_ws, denom_ws, counts_ws, hsem_ws, comb_ws, Nn);
  }

  const int etiles  = (Enum + 15) / 16;
  const int eblocks = (etiles + 1) / 2;
  k_edge_a<<<eblocks, 64, 0, stream>>>(h, x, idx_i, idx_j, b_edge_in, b_edge_h,
                                       b_edge_out, W_att, b_att, f_ein, f_eh, f_eout,
                                       hedge_ws, logit_ws, segmax_ws, Enum);

  k_edge_b<<<(Enum + 255) / 256, 256, 0, stream>>>(idx_i, segmax_ws, logit_ws,
                                                   denom_ws, counts_ws, Enum);

  k_edge_c<<<eblocks, 64, 0, stream>>>(x, idx_i, idx_j, hedge_ws, logit_ws,
                                       denom_ws, f_xmix, hsem_ws, comb_ws, Enum);

  const int ntiles  = (Nn + 15) / 16;
  const int nblocks = (ntiles + 1) / 2;
  k_node<<<nblocks, 64, 0, stream>>>(h, x, v, counts_ws, hsem_ws, comb_ws,
                                     b_node1, b_node2, b_post1, b_post2, b_vel1,
                                     W_vel2, w_v_mix, f_p1, f_p2, f_n1, f_n2, f_v1,
                                     (float*)d_out, Nn);
}
